// Dual_Quantize7_43645457662427
// MI455X (gfx1250) — matrix operand order repaired and checked by reading, not compiled, not run on hardware
//
#include <hip/hip_runtime.h>
#include <hip/hip_bf16.h>
#include <stdint.h>

// ---------------------------------------------------------------------------
// Problem constants (from reference): B=8,H=W=64,dim=256,n_embed=1024
// ---------------------------------------------------------------------------
#define DIM     256
#define NE      1024
#define NROW    32768            // rows per input (8*64*64)
#define QSZ     8388608ull       // NROW*DIM
#define DN      33554432ull      // NROW*NE
#define O_DIFF  33554432ull      // after 4 q tensors
#define O_E     33554436ull      // after 4 diffs
#define O_DIST  33685508ull      // after 4 e index arrays (4*32768)

// ws layout: floats [0,1024) code norms; [1024,66560) row norms (hr,lr);
//            then bf16 embedT[1024][256] at ushort offset 133120 (512 KB)
#define WS_CN    0
#define WS_XN    1024
#define WS_EBT_U16 133120        // = 66560 floats * 2 ushorts/float

typedef __bf16 bf16_t;
typedef bf16_t v16bf __attribute__((ext_vector_type(16)));
typedef float  v8f   __attribute__((ext_vector_type(8)));

__device__ __forceinline__ unsigned short f2bf(float f) {
  union { float f; uint32_t u; } v; v.f = f;
  uint32_t u = v.u;
  u += 0x7FFFu + ((u >> 16) & 1u);   // round-to-nearest-even
  return (unsigned short)(u >> 16);
}

// packed f32x2 -> bf16x2 (hardware cvt when the builtin exists)
__device__ __forceinline__ uint32_t f2bf2(float lo, float hi) {
#if __has_builtin(__builtin_amdgcn_cvt_pk_bf16_f32)
  typedef __bf16 v2bf __attribute__((ext_vector_type(2)));
  union { v2bf v; uint32_t u; } c;
  c.v = __builtin_amdgcn_cvt_pk_bf16_f32(lo, hi);
  return c.u;
#else
  return (uint32_t)f2bf(lo) | ((uint32_t)f2bf(hi) << 16);
#endif
}

// ---------------------------------------------------------------------------
// Kernel 0: norms + zero diffs + pre-transposed bf16 codebook embedT[j][k]
// ---------------------------------------------------------------------------
__global__ __launch_bounds__(256) void vq_prep_kernel(
    const float* __restrict__ xhr, const float* __restrict__ xlr,
    const float* __restrict__ embed, float* __restrict__ ws,
    float* __restrict__ out)
{
  int tid = blockIdx.x * 256 + threadIdx.x;
  if (tid < 4) out[O_DIFF + tid] = 0.0f;
  if (tid < NE) {
    float s = 0.0f;
    for (int k = 0; k < DIM; ++k) {
      float e = embed[(size_t)k * NE + tid];
      s += e * e;
    }
    ws[WS_CN + tid] = s;
  } else if (tid < NE + 2 * NROW) {
    int r = tid - NE;
    const float* X = (r < NROW ? xhr : xlr) + (size_t)(r & (NROW - 1)) * DIM;
    float s = 0.0f;
    for (int k = 0; k < DIM; k += 4) {
      float4 v = *reinterpret_cast<const float4*>(X + k);
      s += v.x * v.x + v.y * v.y + v.z * v.z + v.w * v.w;
    }
    ws[WS_XN + r] = s;
  } else if (tid < NE + 2 * NROW + DIM * NE) {
    int q = tid - (NE + 2 * NROW);      // 0 .. 262143
    int j = q & (NE - 1);               // coalesced read along j
    int k = q >> 10;                    // 0..255
    unsigned short* ebt = reinterpret_cast<unsigned short*>(ws) + WS_EBT_U16;
    ebt[(size_t)j * DIM + k] = f2bf(embed[(size_t)k * NE + j]);
  }
}

// ---------------------------------------------------------------------------
// Kernel 1: dist = ||x||^2 - 2 x.E + ||e||^2 via WMMA bf16 GEMM
//   grid (8, 256, 2): x = N tile (128 cols), y = M tile (128 rows), z = hr/lr
//   8 waves/block; wave tile 32(M) x 64(N) -> 2x4 v8f accumulators.
//   Full 128x256 A tile staged once in LDS (bf16); B fragments loaded
//   directly from the pre-transposed L2-resident bf16 codebook -> the K loop
//   has no barriers and no conversions.
// ---------------------------------------------------------------------------
#define ASTRIDE 264   // bf16 elems per LDS row (528B, 16B-aligned, padded)

__global__ __launch_bounds__(256) void vq_dist_wmma_kernel(
    const float* __restrict__ xhr, const float* __restrict__ xlr,
    const float* __restrict__ ws_f, float* __restrict__ out)
{
  __shared__ unsigned short As[128 * ASTRIDE];   // 66 KB: A[row][k] full-K

  const int t    = threadIdx.x;
  const int lane = t & 31;
  const int wid  = t >> 5;
  const int wm   = wid & 3;      // M offset wm*32
  const int wn   = wid >> 2;     // N offset wn*64
  const int z    = blockIdx.z;
  const float* X  = z ? xlr : xhr;
  const float* xn = ws_f + WS_XN + (size_t)z * NROW;
  const float* cn = ws_f + WS_CN;
  const unsigned short* ebt =
      reinterpret_cast<const unsigned short*>(ws_f) + WS_EBT_U16;
  const int M0 = blockIdx.y * 128;
  const int N0 = blockIdx.x * 128;

  // ---- stage entire A tile (128 rows x 256 k), f32 -> bf16, once ----
  #pragma unroll
  for (int p = 0; p < 32; ++p) {
    int q   = t + p * 256;
    int row = q >> 6;        // 0..127
    int seg = q & 63;        // float4 segment within the row
    float4 v = *reinterpret_cast<const float4*>(
        X + (size_t)(M0 + row) * DIM + seg * 4);
    uint32_t lo = f2bf2(v.x, v.y);
    uint32_t hi = f2bf2(v.z, v.w);
    uint64_t pk = ((uint64_t)hi << 32) | lo;
    *reinterpret_cast<uint64_t*>(&As[row * ASTRIDE + seg * 4]) = pk;
  }
  __syncthreads();

  v8f acc[2][4];
  #pragma unroll
  for (int mi = 0; mi < 2; ++mi)
    #pragma unroll
    for (int ni = 0; ni < 4; ++ni)
      #pragma unroll
      for (int r = 0; r < 8; ++r) acc[mi][ni][r] = 0.0f;

  const int lh = lane >> 4;    // lane half (0/1)
  const int ll = lane & 15;

  #pragma unroll
  for (int kt = 0; kt < 8; ++kt) {
    const int K0 = kt * 32;
    // ---- fragments (per ISA 16-bit A 16x32 / B 32x16 VGPR layouts) ----
    union Frag { v16bf v; uint4 q[2]; } af[2], bfr[4];
    #pragma unroll
    for (int mi = 0; mi < 2; ++mi) {
      const unsigned short* base =
          &As[(wm * 32 + mi * 16 + ll) * ASTRIDE + K0];
      af[mi].q[0] = *reinterpret_cast<const uint4*>(base + lh * 8);       // K 0..7 / 8..15
      af[mi].q[1] = *reinterpret_cast<const uint4*>(base + 16 + lh * 8);  // K 16..23 / 24..31
    }
    #pragma unroll
    for (int ni = 0; ni < 4; ++ni) {
      const unsigned short* base =
          ebt + (size_t)(N0 + wn * 64 + ni * 16 + ll) * DIM + K0;
      bfr[ni].q[0] = *reinterpret_cast<const uint4*>(base + lh * 8);       // K 0..7 / 8..15
      bfr[ni].q[1] = *reinterpret_cast<const uint4*>(base + 16 + lh * 8);  // K 16..23 / 24..31
      if (kt < 7) __builtin_prefetch(base + 32, 0, 1);            // next K-step
    }
    #pragma unroll
    for (int mi = 0; mi < 2; ++mi)
      #pragma unroll
      for (int ni = 0; ni < 4; ++ni)
        acc[mi][ni] = __builtin_amdgcn_wmma_f32_16x16x32_bf16(
            false, af[mi].v, false, bfr[ni].v, (short)0, acc[mi][ni],
            false, false);
  }

  // ---- epilogue: dist = xn - 2*acc + cn, write both duplicated copies ----
  const size_t d1 = O_DIST + (size_t)z * DN;
  const size_t d2 = d1 + 2ull * DN;
  #pragma unroll
  for (int mi = 0; mi < 2; ++mi) {
    #pragma unroll
    for (int ni = 0; ni < 4; ++ni) {
      int colg   = N0 + wn * 64 + ni * 16 + ll;
      float cnv  = cn[colg];
      #pragma unroll
      for (int r = 0; r < 8; ++r) {
        int rowg = M0 + wm * 32 + mi * 16 + r + lh * 8;
        float dv = xn[rowg] - 2.0f * acc[mi][ni][r] + cnv;
        size_t off = (size_t)rowg * NE + colg;
        out[d1 + off] = dv;
        out[d2 + off] = dv;
      }
    }
  }
}

// ---------------------------------------------------------------------------
// Kernel 2: per-row argmin (hc) + top-64 categorical sample (lc)
//   one wave32 per row; 32 p-values per lane in registers
// ---------------------------------------------------------------------------
__global__ __launch_bounds__(256) void vq_sample_kernel(float* __restrict__ out)
{
  __shared__ float pv[8][64];
  __shared__ int   pi[8][64];
  const int lane = threadIdx.x & 31;
  const int wid  = threadIdx.x >> 5;
  const int row  = blockIdx.x * 8 + wid;     // 0..65535
  const int z    = row >> 15;                // 0=hr, 1=lr
  const int r    = row & (NROW - 1);
  const float* dist = out + O_DIST + (size_t)z * DN + (size_t)r * NE;

  float p[32];
  #pragma unroll
  for (int i = 0; i < 8; ++i) {
    float4 d = *reinterpret_cast<const float4*>(dist + i * 128 + lane * 4);
    p[i * 4 + 0] = 1.0f / d.x;
    p[i * 4 + 1] = 1.0f / d.y;
    p[i * 4 + 2] = 1.0f / d.z;
    p[i * 4 + 3] = 1.0f / d.w;
  }
  // lane-local running max
  float lm = -3.0e38f; int ls = 0;
  #pragma unroll
  for (int s = 0; s < 32; ++s)
    if (p[s] > lm) { lm = p[s]; ls = s; }

  float S = 0.0f;
  for (int it = 0; it < 64; ++it) {
    float v = lm; int wl = lane;
    #pragma unroll
    for (int off = 16; off > 0; off >>= 1) {
      float ov = __shfl_xor(v, off);
      int   ol = __shfl_xor(wl, off);
      if (ov > v || (ov == v && ol < wl)) { v = ov; wl = ol; }
    }
    S += v;                                   // identical in all lanes
    if (lane == wl) {                         // winner extracts its element
      int col = (ls >> 2) * 128 + lane * 4 + (ls & 3);
      pv[wid][it] = v;
      pi[wid][it] = col;
      p[ls] = -3.0e38f;
      lm = -3.0e38f; ls = 0;
      #pragma unroll
      for (int s2 = 0; s2 < 32; ++s2)
        if (p[s2] > lm) { lm = p[s2]; ls = s2; }
    }
  }

  if (lane == 0) {
    int hc = pi[wid][0];                      // max p == argmin dist
    // deterministic uniform (splitmix64 of row id)
    uint64_t x = (uint64_t)(row + 1) * 0x9E3779B97F4A7C15ull;
    x ^= x >> 30; x *= 0xBF58476D1CE4E5B9ull;
    x ^= x >> 27; x *= 0x94D049BB133111EBull;
    x ^= x >> 31;
    float u = (float)(x >> 40) * (1.0f / 16777216.0f);
    float target = u * S;
    int lc = pi[wid][63];
    float cum = 0.0f;
    for (int it = 0; it < 64; ++it) {
      cum += pv[wid][it];
      if (cum >= target) { lc = pi[wid][it]; break; }
    }
    out[O_E + (size_t)z * NROW + r]                  = (float)hc;  // e_hc_*
    out[O_E + 2ull * NROW + (size_t)z * NROW + r]    = (float)lc;  // e_lc_*
  }
}

// ---------------------------------------------------------------------------
// Kernel 3: gather codes -> q outputs (straight-through == code values),
//           accumulate commitment MSEs
// ---------------------------------------------------------------------------
__global__ __launch_bounds__(256) void vq_gather_kernel(
    const float* __restrict__ xhr, const float* __restrict__ xlr,
    const float* __restrict__ embed, float* __restrict__ out)
{
  const int lane = threadIdx.x & 31;
  const int wid  = threadIdx.x >> 5;
  const int row  = blockIdx.x * 8 + wid;
  const int z    = row >> 15;
  const int r    = row & (NROW - 1);
  const float* X = (z ? xlr : xhr) + (size_t)r * DIM;
  const int ihc = (int)out[O_E + (size_t)z * NROW + r];
  const int ilc = (int)out[O_E + 2ull * NROW + (size_t)z * NROW + r];
  float* qhc = out + (size_t)z * QSZ + (size_t)r * DIM;              // q_hc_*
  float* qlc = out + 2ull * QSZ + (size_t)z * QSZ + (size_t)r * DIM; // q_lc_*

  float shc = 0.0f, slc = 0.0f;
  #pragma unroll
  for (int j = 0; j < 8; ++j) {
    int k = j * 32 + lane;
    float xv  = X[k];
    float chv = embed[(size_t)k * NE + ihc];   // codes[ihc][k] == embed[k][ihc]
    float clv = embed[(size_t)k * NE + ilc];
    qhc[k] = chv;
    qlc[k] = clv;
    float dh = chv - xv, dl = clv - xv;
    shc += dh * dh;
    slc += dl * dl;
  }
  #pragma unroll
  for (int off = 16; off > 0; off >>= 1) {
    shc += __shfl_xor(shc, off);
    slc += __shfl_xor(slc, off);
  }
  if (lane == 0) {
    const float inv = 1.0f / 8388608.0f;       // mean over B*H*W*dim
    atomicAdd(out + O_DIFF + z,       shc * inv);   // diff_hc_hr / diff_hc_lr
    atomicAdd(out + O_DIFF + 2 + z,   slc * inv);   // diff_lc_hr / diff_lc_lr
  }
}

// ---------------------------------------------------------------------------
extern "C" void kernel_launch(void* const* d_in, const int* in_sizes, int n_in,
                              void* d_out, int out_size, void* d_ws, size_t ws_size,
                              hipStream_t stream) {
  const float* xhr   = (const float*)d_in[0];
  const float* xlr   = (const float*)d_in[1];
  const float* embed = (const float*)d_in[2];
  float* out = (float*)d_out;
  float* ws  = (float*)d_ws;   // ~780 KB used (norms + bf16 embedT)

  // tasks: 4 diffs + 1024 code norms + 65536 row norms + 262144 transposes
  vq_prep_kernel<<<1284, 256, 0, stream>>>(xhr, xlr, embed, ws, out);
  vq_dist_wmma_kernel<<<dim3(8, 256, 2), 256, 0, stream>>>(xhr, xlr, ws, out);
  vq_sample_kernel<<<8192, 256, 0, stream>>>(out);
  vq_gather_kernel<<<8192, 256, 0, stream>>>(xhr, xlr, embed, out);
}
